// Transformer_77137612636583
// MI455X (gfx1250) — matrix operand order repaired and checked by reading, not compiled, not run on hardware
//
#include <hip/hip_runtime.h>
#include <math.h>
#include <stdint.h>

// ---------------- types & constants ----------------
typedef __attribute__((ext_vector_type(16))) __bf16 v16bf;
typedef __attribute__((ext_vector_type(8)))  __bf16 v8bf;
typedef __attribute__((ext_vector_type(8)))  float  v8f;
typedef unsigned int u32;
typedef unsigned long long u64;
typedef __attribute__((ext_vector_type(4))) u32 su4;
typedef __attribute__((ext_vector_type(8))) u32 su8;

#define DEPTH 4
#define HEADS 8
#define DHEAD 64
#define DIMN  512
#define MLPN  2048
#define REDN  128
#define BB 8
#define PP 4
#define NN 256
#define TOK (BB*PP*NN)          /* 8192 tokens */
#define EQKV (3*HEADS*DHEAD)    /* 1536 */

__device__ __forceinline__ float gelu_f(float x) {
  return 0.5f * x * (1.0f + erff(x * 0.70710678118654752f));
}
__device__ __forceinline__ float sigmoid_f(float x) {
  return 1.0f / (1.0f + expf(-x));
}
__device__ __forceinline__ v8f wmma_bf16(v16bf a, v16bf b, v8f c) {
  return __builtin_amdgcn_wmma_f32_16x16x32_bf16(false, a, false, b, (short)0, c,
                                                 false, false);
}
// k order this file uses for A and B (contiguous): VGPR i holds a K pair; lanes 0-15 K base 0, 16-31 +16
__device__ __forceinline__ int a_koff(int i, int half) {
  return 2 * i + 16 * half;
}

// ---------------- CDNA5 data movers ----------------
// async global -> LDS, 16 bytes, tracked by ASYNCcnt
__device__ __forceinline__ void async_ld_b128(const void* g, void* lds) {
  u32 loff = (u32)(u64)(uintptr_t)lds;  // low 32 bits of generic ptr = LDS byte addr
  asm volatile("global_load_async_to_lds_b128 %0, %1, off"
               :: "v"(loff), "v"(g) : "memory");
}
__device__ __forceinline__ void wait_async0() {
  asm volatile("s_wait_asynccnt 0x0" ::: "memory");
}
// TDM: 2D tile (tile_d1 rows x tile_d0 elems, 2-byte elems, row stride in elems)
__device__ __forceinline__ void tdm_load_2d(const void* gptr, void* ldsptr,
                                            u32 tile_d0, u32 tile_d1, u32 stride_el) {
  u64 ga = (u64)(uintptr_t)gptr;
  u32 la = (u32)(uintptr_t)ldsptr;
  su4 g0;
  g0[0] = 1u;                                             // count=1, user mode
  g0[1] = la;                                             // lds_addr
  g0[2] = (u32)ga;                                        // global_addr[31:0]
  g0[3] = ((u32)(ga >> 32) & 0x01FFFFFFu) | (2u << 30);   // addr[56:32] | type=2
  const u32 dim = 0x100000u;                              // huge tensor dims: no OOB
  su8 g1;
  g1[0] = (1u << 16);                                     // data_size = 2 bytes
  g1[1] = (dim & 0xFFFFu) << 16;                          // tensor_dim0 lo
  g1[2] = (dim >> 16) | ((dim & 0xFFFFu) << 16);          // dim0 hi | dim1 lo
  g1[3] = (dim >> 16) | (tile_d0 << 16);                  // dim1 hi | tile_dim0
  g1[4] = tile_d1;                                        // tile_dim1 (tile_dim2=0)
  g1[5] = stride_el;                                      // tensor_dim0_stride lo
  g1[6] = 0u;
  g1[7] = 0u;
  asm volatile("tensor_load_to_lds %0, %1" :: "s"(g0), "s"(g1) : "memory");
}

// ---------------- elementwise helpers ----------------
__global__ void k_copy_f32(const float* __restrict__ in, float* __restrict__ out, int n) {
  int i = blockIdx.x * 256 + threadIdx.x;
  if (i < n) out[i] = in[i];
}
__global__ void k_f32_to_bf16(const float* __restrict__ in, __bf16* __restrict__ out, int n) {
  int i = blockIdx.x * 256 + threadIdx.x;
  if (i < n) out[i] = (__bf16)in[i];
}

// ---------------- layernorm (row = 512) ----------------
__global__ void k_layernorm(const float* __restrict__ x, const float* __restrict__ g,
                            const float* __restrict__ b, __bf16* __restrict__ out) {
  __shared__ float ssum[128], ssq[128];
  int row = blockIdx.x, tid = threadIdx.x;
  const float* xr = x + (size_t)row * DIMN;
  float v0 = xr[tid], v1 = xr[tid + 128], v2 = xr[tid + 256], v3 = xr[tid + 384];
  ssum[tid] = v0 + v1 + v2 + v3;
  ssq[tid] = v0 * v0 + v1 * v1 + v2 * v2 + v3 * v3;
  __syncthreads();
  for (int st = 64; st > 0; st >>= 1) {
    if (tid < st) { ssum[tid] += ssum[tid + st]; ssq[tid] += ssq[tid + st]; }
    __syncthreads();
  }
  float mean = ssum[0] * (1.0f / DIMN);
  float var = ssq[0] * (1.0f / DIMN) - mean * mean;
  float inv = rsqrtf(var + 1e-5f);
  __bf16* orow = out + (size_t)row * DIMN;
  orow[tid]       = (__bf16)((v0 - mean) * inv * g[tid]       + b[tid]);
  orow[tid + 128] = (__bf16)((v1 - mean) * inv * g[tid + 128] + b[tid + 128]);
  orow[tid + 256] = (__bf16)((v2 - mean) * inv * g[tid + 256] + b[tid + 256]);
  orow[tid + 384] = (__bf16)((v3 - mean) * inv * g[tid + 384] + b[tid + 384]);
}

// ---------------- generic bf16 WMMA GEMM ----------------
// C[M,N] = act(A @ B + bias); omode 0: bf16 out, 1: f32 out, 2: f32 out += residual
// block = 128 (4 waves in 2x2; each wave computes a 32x32 macro-tile = 4 WMMA/K-step)
// grid = (N/64, M/64). A tile (64x32) staged by TDM; B tile (32x64) transposed by VALU.
__global__ void k_gemm(const __bf16* __restrict__ A, const __bf16* __restrict__ B,
                       const float* __restrict__ bias, const float* __restrict__ residual,
                       float* __restrict__ outF, __bf16* __restrict__ outB,
                       int M, int N, int K, int act, int omode) {
  __shared__ __bf16 As[64 * 32];   // row-major [m][k]   4 KB
  __shared__ __bf16 BsT[64 * 32];  // transposed [n][k]  4 KB
  int tid = threadIdx.x;
  int lane = tid & 31, wave = tid >> 5;
  int wr = wave >> 1, wc = wave & 1;
  int l16 = lane & 15, half = lane >> 4;
  int m0 = blockIdx.y * 64, n0 = blockIdx.x * 64;

  v8f acc[2][2];
#pragma unroll
  for (int sr = 0; sr < 2; sr++)
#pragma unroll
    for (int sc = 0; sc < 2; sc++)
      acc[sr][sc] = {0.f, 0.f, 0.f, 0.f, 0.f, 0.f, 0.f, 0.f};

  for (int k0 = 0; k0 < K; k0 += 32) {
    if (wave == 0)  // TDM DMA: A tile 64 rows x 32 elems, row stride K elems
      tdm_load_2d(A + (size_t)m0 * K + k0, As, 32u, 64u, (u32)K);
    // B tile 32(k) x 64(n): each thread 2 chunks of 8 bf16, scatter into BsT[n][k]
#pragma unroll
    for (int j = 0; j < 2; j++) {
      int c = tid + 128 * j;
      int krow = c >> 3, ngrp = c & 7;
      v8bf bvec = *(const v8bf*)(B + (size_t)(k0 + krow) * N + n0 + ngrp * 8);
#pragma unroll
      for (int e = 0; e < 8; e++) BsT[(ngrp * 8 + e) * 32 + krow] = bvec[e];
      if (k0 + 32 < K)
        __builtin_prefetch(B + (size_t)(k0 + 32 + krow) * N + n0 + ngrp * 8, 0, 1);
    }
    if (wave == 0) __builtin_amdgcn_s_wait_tensorcnt(0);
    __syncthreads();

    v16bf a[2], b[2];
#pragma unroll
    for (int s = 0; s < 2; s++) {
      const __bf16* ap = As + (wr * 32 + s * 16 + l16) * 32;
      const __bf16* bp = BsT + (wc * 32 + s * 16 + l16) * 32 + 16 * half;
#pragma unroll
      for (int i = 0; i < 8; i++) {
        int ka = a_koff(i, half);
        a[s][2 * i]     = ap[ka];
        a[s][2 * i + 1] = ap[ka + 1];
        b[s][2 * i]     = bp[2 * i];
        b[s][2 * i + 1] = bp[2 * i + 1];
      }
    }
#pragma unroll
    for (int sr = 0; sr < 2; sr++)
#pragma unroll
      for (int sc = 0; sc < 2; sc++)
        acc[sr][sc] = wmma_bf16(a[sr], b[sc], acc[sr][sc]);
    __syncthreads();
  }

#pragma unroll
  for (int sr = 0; sr < 2; sr++)
#pragma unroll
    for (int sc = 0; sc < 2; sc++) {
      int col = n0 + wc * 32 + sc * 16 + l16;
      float bv = bias ? bias[col] : 0.0f;
#pragma unroll
      for (int r = 0; r < 8; r++) {
        int rowm = m0 + wr * 32 + sr * 16 + r + 8 * half;
        float v = acc[sr][sc][r] + bv;
        if (act == 1) v = gelu_f(v);
        size_t idx = (size_t)rowm * N + col;
        if (omode == 0) {
          outB[idx] = (__bf16)v;
        } else {
          float o = v;
          if (omode == 2) o += residual[idx];
          outF[idx] = o;
        }
      }
    }
}

// ---------------- fused attention ----------------
// grid = B*P*HEADS = 256, block = 128 (4 waves)
// Stage K (async->LDS) and V^T once per (b,p,h), then loop 16-row query chunks.
__global__ void k_attention(const __bf16* __restrict__ qkv, __bf16* __restrict__ att) {
  __shared__ __bf16 Ks[NN * DHEAD];   // [token][d]   32 KB
  __shared__ __bf16 Vt[DHEAD * NN];   // [d][token]   32 KB
  __shared__ float sc[16 * 256];      // scores       16 KB
  __shared__ __bf16 pr[16 * 256];     // probs         8 KB
  __shared__ float red[16 * 8], rowmax[16], rowsum[16];

  int bph = blockIdx.x;
  int h = bph & 7;
  int bp = bph >> 3;          // b*PP + p
  int tb = bp * NN;           // first token of this sequence
  int tid = threadIdx.x, lane = tid & 31, wave = tid >> 5;
  int l16 = lane & 15, half = lane >> 4;

  // ---- stage K via async global->LDS (ASYNCcnt), V transposed via VALU ----
  for (int j = 0; j < 16; j++) {
    int c = tid + 128 * j;          // 2048 chunks of 8 bf16
    int tok = c >> 3, d8 = c & 7;
    async_ld_b128(qkv + (size_t)(tb + tok) * EQKV + DIMN + h * DHEAD + d8 * 8,
                  Ks + tok * DHEAD + d8 * 8);
    v8bf vv = *(const v8bf*)(qkv + (size_t)(tb + tok) * EQKV + 2 * DIMN + h * DHEAD + d8 * 8);
#pragma unroll
    for (int e = 0; e < 8; e++) Vt[(d8 * 8 + e) * NN + tok] = vv[e];
  }
  wait_async0();
  __syncthreads();

  const float scale = 0.125f;  // 64^-0.5
  for (int rt = 0; rt < 16; rt++) {
    int rb = tb + rt * 16;

    // ---- phase 1: S = Q K^T; wave covers 64 key columns ----
    v8f accs[4];
#pragma unroll
    for (int t = 0; t < 4; t++) accs[t] = {0.f, 0.f, 0.f, 0.f, 0.f, 0.f, 0.f, 0.f};
    for (int kk = 0; kk < DHEAD; kk += 32) {
      v16bf a;
      const __bf16* qp = qkv + (size_t)(rb + l16) * EQKV + h * DHEAD + kk;
#pragma unroll
      for (int i = 0; i < 8; i++) {
        int ka = a_koff(i, half);
        a[2 * i] = qp[ka];
        a[2 * i + 1] = qp[ka + 1];
      }
#pragma unroll
      for (int t = 0; t < 4; t++) {
        int cb = wave * 64 + t * 16;
        const __bf16* kp = Ks + (cb + l16) * DHEAD + kk + 16 * half;
        v16bf b;
#pragma unroll
        for (int i = 0; i < 8; i++) {
          b[2 * i] = kp[2 * i];
          b[2 * i + 1] = kp[2 * i + 1];
        }
        accs[t] = wmma_bf16(a, b, accs[t]);
      }
    }
#pragma unroll
    for (int t = 0; t < 4; t++) {
      int cb = wave * 64 + t * 16;
#pragma unroll
      for (int r = 0; r < 8; r++)
        sc[(r + 8 * half) * 256 + cb + l16] = accs[t][r] * scale;
    }
    __syncthreads();

    // ---- phase 2: row softmax (16 rows x 256) ----
    int srow = tid >> 3, seg = tid & 7;
    float mx = -1e30f;
    for (int c = seg * 32; c < seg * 32 + 32; c++) mx = fmaxf(mx, sc[srow * 256 + c]);
    red[srow * 8 + seg] = mx;
    __syncthreads();
    if (seg == 0) {
      float m = red[srow * 8];
      for (int j = 1; j < 8; j++) m = fmaxf(m, red[srow * 8 + j]);
      rowmax[srow] = m;
    }
    __syncthreads();
    float rm = rowmax[srow], s = 0.f;
    for (int c = seg * 32; c < seg * 32 + 32; c++) {
      float e = expf(sc[srow * 256 + c] - rm);
      sc[srow * 256 + c] = e;
      s += e;
    }
    red[srow * 8 + seg] = s;
    __syncthreads();
    if (seg == 0) {
      float t2 = 0.f;
      for (int j = 0; j < 8; j++) t2 += red[srow * 8 + j];
      rowsum[srow] = t2;
    }
    __syncthreads();
    float inv = 1.0f / rowsum[srow];
    for (int c = seg * 32; c < seg * 32 + 32; c++)
      pr[srow * 256 + c] = (__bf16)(sc[srow * 256 + c] * inv);
    __syncthreads();

    // ---- phase 3: O = P V; wave covers 16 of 64 head dims ----
    v8f oacc = {0.f, 0.f, 0.f, 0.f, 0.f, 0.f, 0.f, 0.f};
    for (int kk = 0; kk < NN; kk += 32) {
      v16bf a, b;
      const __bf16* vp = Vt + (wave * 16 + l16) * NN + kk + 16 * half;
#pragma unroll
      for (int i = 0; i < 8; i++) {
        int ka = a_koff(i, half);
        a[2 * i] = pr[l16 * 256 + kk + ka];
        a[2 * i + 1] = pr[l16 * 256 + kk + ka + 1];
        b[2 * i] = vp[2 * i];
        b[2 * i + 1] = vp[2 * i + 1];
      }
      oacc = wmma_bf16(a, b, oacc);
    }
#pragma unroll
    for (int r = 0; r < 8; r++) {
      int rowtok = rb + r + 8 * half;
      att[(size_t)rowtok * DIMN + h * DHEAD + wave * 16 + l16] = (__bf16)oacc[r];
    }
    __syncthreads();  // pr/sc free before next chunk
  }
}

// ---------------- BiAttn pieces ----------------
__global__ void k_pool(const __bf16* __restrict__ yb, float* __restrict__ xg) {
  int bn = blockIdx.x;
  int b = bn >> 8, n = bn & 255;
  for (int d = threadIdx.x; d < DIMN; d += 128) {
    float s = 0.f;
#pragma unroll
    for (int p = 0; p < PP; p++)
      s += (float)yb[(size_t)((b * PP + p) * NN + n) * DIMN + d];
    xg[(size_t)bn * DIMN + d] = s * 0.25f;
  }
}

__global__ void k_small_mm(const float* __restrict__ A, const float* __restrict__ W,
                           const float* __restrict__ bias, float* __restrict__ out,
                           int K, int N, int act) {
  int row = blockIdx.x;
  for (int c = threadIdx.x; c < N; c += blockDim.x) {
    float s = bias[c];
    for (int k = 0; k < K; k++) s += A[(size_t)row * K + k] * W[(size_t)k * N + c];
    s = (act == 1) ? gelu_f(s) : sigmoid_f(s);
    out[(size_t)row * N + c] = s;
  }
}

__global__ void k_biattn_final(const float* __restrict__ y, const float* __restrict__ xl,
                               const float* __restrict__ g, const float* __restrict__ catt,
                               const float* __restrict__ wsv, const float* __restrict__ bs,
                               float* __restrict__ x) {
  __shared__ float red2[128];
  __shared__ float s_s;
  int t = blockIdx.x;
  int b = t >> 10;
  int n = t & 255;
  int bn = b * NN + n;
  int r = threadIdx.x;  // 128 == RED
  red2[r] = xl[(size_t)t * REDN + r] * wsv[r] + g[(size_t)bn * REDN + r] * wsv[REDN + r];
  __syncthreads();
  for (int st = 64; st > 0; st >>= 1) {
    if (r < st) red2[r] += red2[r + st];
    __syncthreads();
  }
  if (r == 0) s_s = sigmoid_f(red2[0] + bs[0]);
  __syncthreads();
  float sa = s_s;
  for (int d = r; d < DIMN; d += 128) {
    size_t idx = (size_t)t * DIMN + d;
    x[idx] += y[idx] * catt[(size_t)bn * DIMN + d] * sa;
  }
}

// ---------------- workspace layout (bytes) ----------------
static const size_t OF_QKV  = 0;
static const size_t OF_H    = OF_QKV + (size_t)TOK * EQKV * 2;
static const size_t OF_Y    = OF_H + (size_t)TOK * MLPN * 2;
static const size_t OF_XB   = OF_Y + (size_t)TOK * DIMN * 4;
static const size_t OF_XG   = OF_XB + (size_t)TOK * DIMN * 2;
static const size_t OF_G    = OF_XG + (size_t)BB * NN * DIMN * 4;
static const size_t OF_CATT = OF_G + (size_t)BB * NN * REDN * 4;
static const size_t OF_XL   = OF_CATT + (size_t)BB * NN * DIMN * 4;
static const size_t OF_WQKV = OF_XL + (size_t)TOK * REDN * 4;
static const size_t OF_WO   = OF_WQKV + (size_t)DIMN * EQKV * 2;
static const size_t OF_W1   = OF_WO + (size_t)DIMN * DIMN * 2;
static const size_t OF_W2   = OF_W1 + (size_t)DIMN * MLPN * 2;
static const size_t OF_WL   = OF_W2 + (size_t)MLPN * DIMN * 2;

extern "C" void kernel_launch(void* const* d_in, const int* in_sizes, int n_in,
                              void* d_out, int out_size, void* d_ws, size_t ws_size,
                              hipStream_t stream) {
  const float* xin  = (const float*)d_in[0];
  const float* ln1g = (const float*)d_in[1];
  const float* ln1b = (const float*)d_in[2];
  const float* wqkv = (const float*)d_in[3];
  const float* wo   = (const float*)d_in[4];
  const float* bo   = (const float*)d_in[5];
  const float* ln2g = (const float*)d_in[6];
  const float* ln2b = (const float*)d_in[7];
  const float* w1   = (const float*)d_in[8];
  const float* b1   = (const float*)d_in[9];
  const float* w2   = (const float*)d_in[10];
  const float* b2   = (const float*)d_in[11];
  const float* bng  = (const float*)d_in[12];
  const float* bnb  = (const float*)d_in[13];
  const float* wg   = (const float*)d_in[14];
  const float* bg   = (const float*)d_in[15];
  const float* wl   = (const float*)d_in[16];
  const float* bl   = (const float*)d_in[17];
  const float* wc   = (const float*)d_in[18];
  const float* bc   = (const float*)d_in[19];
  const float* wsv  = (const float*)d_in[20];
  const float* bsv  = (const float*)d_in[21];

  char* wsb = (char*)d_ws;
  __bf16* qkvB = (__bf16*)(wsb + OF_QKV);
  __bf16* hB   = (__bf16*)(wsb + OF_H);
  float*  yF   = (float*)(wsb + OF_Y);
  __bf16* xB   = (__bf16*)(wsb + OF_XB);
  float*  xgF  = (float*)(wsb + OF_XG);
  float*  gF   = (float*)(wsb + OF_G);
  float*  cattF= (float*)(wsb + OF_CATT);
  float*  xlF  = (float*)(wsb + OF_XL);
  __bf16* wqkvB= (__bf16*)(wsb + OF_WQKV);
  __bf16* woB  = (__bf16*)(wsb + OF_WO);
  __bf16* w1B  = (__bf16*)(wsb + OF_W1);
  __bf16* w2B  = (__bf16*)(wsb + OF_W2);
  __bf16* wlB  = (__bf16*)(wsb + OF_WL);

  float* x = (float*)d_out;

  k_copy_f32<<<(TOK * DIMN + 255) / 256, 256, 0, stream>>>(xin, x, TOK * DIMN);

  for (int i = 0; i < DEPTH; i++) {
    const float* Lln1g = ln1g + i * DIMN;
    const float* Lln1b = ln1b + i * DIMN;
    const float* Lwqkv = wqkv + (size_t)i * DIMN * EQKV;
    const float* Lwo   = wo + (size_t)i * DIMN * DIMN;
    const float* Lbo   = bo + i * DIMN;
    const float* Lln2g = ln2g + i * DIMN;
    const float* Lln2b = ln2b + i * DIMN;
    const float* Lw1   = w1 + (size_t)i * DIMN * MLPN;
    const float* Lb1   = b1 + i * MLPN;
    const float* Lw2   = w2 + (size_t)i * MLPN * DIMN;
    const float* Lb2   = b2 + i * DIMN;
    const float* Lbng  = bng + i * DIMN;
    const float* Lbnb  = bnb + i * DIMN;
    const float* Lwg   = wg + (size_t)i * DIMN * REDN;
    const float* Lbg   = bg + i * REDN;
    const float* Lwl   = wl + (size_t)i * DIMN * REDN;
    const float* Lbl   = bl + i * REDN;
    const float* Lwc   = wc + (size_t)i * REDN * DIMN;
    const float* Lbc   = bc + i * DIMN;
    const float* Lws   = wsv + (size_t)i * 2 * REDN;
    const float* Lbs   = bsv + i;

    k_f32_to_bf16<<<(DIMN * EQKV + 255) / 256, 256, 0, stream>>>(Lwqkv, wqkvB, DIMN * EQKV);
    k_f32_to_bf16<<<(DIMN * DIMN + 255) / 256, 256, 0, stream>>>(Lwo, woB, DIMN * DIMN);
    k_f32_to_bf16<<<(DIMN * MLPN + 255) / 256, 256, 0, stream>>>(Lw1, w1B, DIMN * MLPN);
    k_f32_to_bf16<<<(MLPN * DIMN + 255) / 256, 256, 0, stream>>>(Lw2, w2B, MLPN * DIMN);
    k_f32_to_bf16<<<(DIMN * REDN + 255) / 256, 256, 0, stream>>>(Lwl, wlB, DIMN * REDN);

    // ---- attention block ----
    k_layernorm<<<TOK, 128, 0, stream>>>(x, Lln1g, Lln1b, xB);
    {
      dim3 g(EQKV / 64, TOK / 64);
      k_gemm<<<g, 128, 0, stream>>>(xB, wqkvB, nullptr, nullptr, nullptr, qkvB,
                                    TOK, EQKV, DIMN, 0, 0);
    }
    k_attention<<<BB * PP * HEADS, 128, 0, stream>>>(qkvB, xB);
    {
      dim3 g(DIMN / 64, TOK / 64);
      k_gemm<<<g, 128, 0, stream>>>(xB, woB, Lbo, x, x, nullptr,
                                    TOK, DIMN, DIMN, 0, 2);
    }

    // ---- MLP ----
    k_layernorm<<<TOK, 128, 0, stream>>>(x, Lln2g, Lln2b, xB);
    {
      dim3 g(MLPN / 64, TOK / 64);
      k_gemm<<<g, 128, 0, stream>>>(xB, w1B, Lb1, nullptr, nullptr, hB,
                                    TOK, MLPN, DIMN, 1, 0);
    }
    {
      dim3 g(DIMN / 64, TOK / 64);
      k_gemm<<<g, 128, 0, stream>>>(hB, w2B, Lb2, nullptr, yF, nullptr,
                                    TOK, DIMN, MLPN, 0, 1);
    }

    // ---- BiAttn ----
    k_layernorm<<<TOK, 128, 0, stream>>>(yF, Lbng, Lbnb, xB);
    k_pool<<<BB * NN, 128, 0, stream>>>(xB, xgF);
    k_small_mm<<<BB * NN, 128, 0, stream>>>(xgF, Lwg, Lbg, gF, DIMN, REDN, 1);
    k_small_mm<<<BB * NN, 256, 0, stream>>>(gF, Lwc, Lbc, cattF, REDN, DIMN, 2);
    {
      dim3 g(REDN / 64, TOK / 64);
      k_gemm<<<g, 128, 0, stream>>>(xB, wlB, Lbl, nullptr, xlF, nullptr,
                                    TOK, REDN, DIMN, 1, 1);
    }
    k_biattn_final<<<TOK, 128, 0, stream>>>(yF, xlF, gF, cattF, Lws, Lbs, x);
  }

  (void)in_sizes; (void)n_in; (void)out_size; (void)ws_size;
}
